// ParallelMultiHeadedAttention_8323646620321
// MI455X (gfx1250) — compile-verified, matrix operand order checked by reading, not run on hardware
//
#include <hip/hip_runtime.h>
#include <hip/hip_bf16.h>

// Problem dims (fixed by reference)
#define BB 2
#define SS 2048
#define DD 2048
#define HH 16
#define HD 128

typedef __attribute__((ext_vector_type(16))) __bf16 v16bf;
typedef __attribute__((ext_vector_type(8)))  __bf16 v8bf;
typedef __attribute__((ext_vector_type(4)))  __bf16 v4bf;
typedef __attribute__((ext_vector_type(8)))  float  v8f;
typedef __attribute__((ext_vector_type(4)))  float  v4f_t;

static __device__ __forceinline__ __bf16 f2bf(float x) {
  union { float f; unsigned u; } v; v.f = x;
  unsigned r = v.u + 0x7FFFu + ((v.u >> 16) & 1u);   // round-to-nearest-even
  unsigned short h = (unsigned short)(r >> 16);
  return __builtin_bit_cast(__bf16, h);
}

// Load one 16x32 (A) or 32x16 (B, via transposed storage) bf16 WMMA fragment.
// CDNA5 layout: lanes 0-15 hold K{0..7,16..23}, lanes 16-31 hold K{8..15,24..31}
// (caller adds kb = (lane>>4)*8 to the base pointer).
static __device__ __forceinline__ v16bf ld_frag(const __bf16* p) {
  v8bf lo = *(const v8bf*)p;
  v8bf hi = *(const v8bf*)(p + 16);
  v16bf r;
#pragma unroll
  for (int i = 0; i < 8; ++i) { r[i] = lo[i]; r[i + 8] = hi[i]; }
  return r;
}

static __device__ __forceinline__ v8f wmma_bf16(v16bf a, v16bf b, v8f c) {
  return __builtin_amdgcn_wmma_f32_16x16x32_bf16(false, a, false, b, (short)0, c,
                                                 false, false);
}

// Transpose-commit a 4(k) x 8(n) chunk into Bt[n][k]-style LDS: 8x ds_store_b64.
static __device__ __forceinline__ void st_tr4x8(__bf16* base /*&Bt[n8*stride + k4]*/,
                                                int stride, v8bf r0, v8bf r1,
                                                v8bf r2, v8bf r3) {
#pragma unroll
  for (int j = 0; j < 8; ++j) {
    v4bf p; p[0] = r0[j]; p[1] = r1[j]; p[2] = r2[j]; p[3] = r3[j];
    *(v4bf*)(base + j * stride) = p;
  }
}

// ---------------------------------------------------------------- conversion
__global__ __launch_bounds__(256)
void cvt_f32_bf16(const float* __restrict__ in, __bf16* __restrict__ out, long n4) {
  long i = (long)blockIdx.x * blockDim.x + threadIdx.x;
  if (i >= n4) return;
  v4f_t v = *(const v4f_t*)(in + i * 4);
  v4bf o;
#pragma unroll
  for (int j = 0; j < 4; ++j) o[j] = f2bf(v[j]);
  *(v4bf*)(out + i * 4) = o;
}

// ------------------------------------------------------------------- GEMM
// C[M,N] = A[M,K](bf16) * B[K,N](bf16) + bias[N]
// Block tile 128(M) x 256(N), Kc = 32, double-buffered LDS.
// 8 waves arranged 2(M) x 4(N); each wave owns a 64x64 tile = 4x4 WMMA accs.
// Per wave per K-step: 16 ds_load_b128 + 10 ds stores -> 16 v_wmma.
template <bool STORE_BF16>
__global__ __launch_bounds__(256)
void gemm_bf16_dbuf(const __bf16* __restrict__ A, const __bf16* __restrict__ Bm,
                    const float* __restrict__ bias, void* __restrict__ outP,
                    int M, int N, int K) {
  __shared__ __bf16 As[2][128 * 40];   // [m][k], pad 8 -> 80B rows (16B aligned)
  __shared__ __bf16 Bt[2][256 * 40];   // [n][k] transposed

  const int tid  = threadIdx.x;
  const int lane = tid & 31;
  const int wave = tid >> 5;
  const int wm = wave & 1;             // wave row (2 x 64)
  const int wn = wave >> 1;            // wave col (4 x 64)
  const int lh = lane & 15;
  const int kb = (lane >> 4) * 8;

  const int n0 = blockIdx.x * 256;
  const int m0 = blockIdx.y * 128;

  // B staging geometry: each thread owns a 4(k) x 8(n) chunk of the 32x256 tile
  const int bk4 = (tid >> 5) * 4;        // k row group: 0,4,...,28
  const int bn8 = (tid & 31) * 8;        // n chunk: 0..248

  v8f acc[4][4];
#pragma unroll
  for (int mi = 0; mi < 4; ++mi)
#pragma unroll
    for (int ni = 0; ni < 4; ++ni)
#pragma unroll
      for (int r = 0; r < 8; ++r) acc[mi][ni][r] = 0.f;

  // ---- prologue: stage tile k0=0 into buffer 0
#pragma unroll
  for (int i = 0; i < 2; ++i) {
    int cid = i * 256 + tid;
    int row = cid >> 2, k8 = (cid & 3) * 8;
    *(v8bf*)&As[0][row * 40 + k8] = *(const v8bf*)&A[(long)(m0 + row) * K + k8];
  }
  {
    v8bf r0 = *(const v8bf*)&Bm[(long)(bk4 + 0) * N + n0 + bn8];
    v8bf r1 = *(const v8bf*)&Bm[(long)(bk4 + 1) * N + n0 + bn8];
    v8bf r2 = *(const v8bf*)&Bm[(long)(bk4 + 2) * N + n0 + bn8];
    v8bf r3 = *(const v8bf*)&Bm[(long)(bk4 + 3) * N + n0 + bn8];
    st_tr4x8(&Bt[0][bn8 * 40 + bk4], 40, r0, r1, r2, r3);
  }
  __syncthreads();

  int cur = 0;
  for (int k0 = 32; k0 < K; k0 += 32) {
    // ---- prefetch next tile into registers (overlaps with WMMAs below)
    v8bf pa[2], pb[4];
#pragma unroll
    for (int i = 0; i < 2; ++i) {
      int cid = i * 256 + tid;
      int row = cid >> 2, k8 = (cid & 3) * 8;
      pa[i] = *(const v8bf*)&A[(long)(m0 + row) * K + k0 + k8];
    }
#pragma unroll
    for (int i = 0; i < 4; ++i)
      pb[i] = *(const v8bf*)&Bm[(long)(k0 + bk4 + i) * N + n0 + bn8];

    // ---- compute on current buffer
    {
      v16bf Af[4];
#pragma unroll
      for (int mi = 0; mi < 4; ++mi)
        Af[mi] = ld_frag(&As[cur][(wm * 64 + mi * 16 + lh) * 40 + kb]);
#pragma unroll
      for (int ni = 0; ni < 4; ++ni) {
        v16bf Bf = ld_frag(&Bt[cur][(wn * 64 + ni * 16 + lh) * 40 + kb]);
#pragma unroll
        for (int mi = 0; mi < 4; ++mi)
          acc[mi][ni] = wmma_bf16(Af[mi], Bf, acc[mi][ni]);
      }
    }

    // ---- commit prefetched tile to the other buffer (no readers there)
    const int nxt = cur ^ 1;
#pragma unroll
    for (int i = 0; i < 2; ++i) {
      int cid = i * 256 + tid;
      int row = cid >> 2, k8 = (cid & 3) * 8;
      *(v8bf*)&As[nxt][row * 40 + k8] = pa[i];
    }
    st_tr4x8(&Bt[nxt][bn8 * 40 + bk4], 40, pb[0], pb[1], pb[2], pb[3]);
    __syncthreads();   // single barrier per K-step
    cur = nxt;
  }

  // ---- final tile
  {
    v16bf Af[4];
#pragma unroll
    for (int mi = 0; mi < 4; ++mi)
      Af[mi] = ld_frag(&As[cur][(wm * 64 + mi * 16 + lh) * 40 + kb]);
#pragma unroll
    for (int ni = 0; ni < 4; ++ni) {
      v16bf Bf = ld_frag(&Bt[cur][(wn * 64 + ni * 16 + lh) * 40 + kb]);
#pragma unroll
      for (int mi = 0; mi < 4; ++mi)
        acc[mi][ni] = wmma_bf16(Af[mi], Bf, acc[mi][ni]);
    }
  }

  // ---- branch-free epilogue; C/D layout: row = r + 8*(lane>=16), col = lane&15
#pragma unroll
  for (int mi = 0; mi < 4; ++mi)
#pragma unroll
    for (int ni = 0; ni < 4; ++ni) {
      const int col = n0 + wn * 64 + ni * 16 + lh;
      const float bv = bias[col];
#pragma unroll
      for (int r = 0; r < 8; ++r) {
        int row = m0 + wm * 64 + mi * 16 + r + ((lane >> 4) << 3);
        float v = acc[mi][ni][r] + bv;
        if constexpr (STORE_BF16)
          ((__bf16*)outP)[(long)row * N + col] = f2bf(v);
        else
          ((float*)outP)[(long)row * N + col] = v;
      }
    }
}

// --------------------------------------------------------------- attention
// Flash attention over qkv[B*S][3*D] (bf16), head layout: h*384 + {q:0,k:128,v:256}.
// Block = 4 waves; each wave owns a 16-row Q block; K/V tiles (64 keys) shared in LDS.
__global__ __launch_bounds__(128)
void attn_flash(const __bf16* __restrict__ qkv, __bf16* __restrict__ ctx) {
  __shared__ __bf16 Kt[64 * 136];     // [key][hd], pad 8 -> 272B row
  __shared__ __bf16 Vt[128 * 72];     // [hd][key] transposed, pad 8
  __shared__ __bf16 Pb[4][16 * 72];   // per-wave P tile [row][key]

  const int tid  = threadIdx.x;
  const int lane = tid & 31;
  const int wave = tid >> 5;
  const int lh = lane & 15;
  const int hb = lane >> 4;
  const int kb = hb * 8;

  const int qt = blockIdx.x;
  const int h  = blockIdx.y;
  const int b  = blockIdx.z;

  const long QS = 3L * DD;
  const long rowQ0 = (long)b * SS + qt * 64 + wave * 16;
  const float rscale = 0.08838834764831845f;  // 1/sqrt(128)

  v16bf Qf[4];
  {
    const __bf16* qbase = qkv + (rowQ0 + lh) * QS + (long)h * (3 * HD);
#pragma unroll
    for (int c = 0; c < 4; ++c) Qf[c] = ld_frag(qbase + c * 32 + kb);
  }

  float mrow[8], lrow[8];
  v8f Of[8];
#pragma unroll
  for (int r = 0; r < 8; ++r) { mrow[r] = -3.0e38f; lrow[r] = 0.f; }
#pragma unroll
  for (int f = 0; f < 8; ++f)
#pragma unroll
    for (int r = 0; r < 8; ++r) Of[f][r] = 0.f;

  for (int kt = 0; kt < SS; kt += 64) {
    const long keyRow0 = (long)b * SS + kt;
    // stage K tile 64x128 row-major (b128 stores)
#pragma unroll
    for (int i = 0; i < 8; ++i) {
      int cid = i * 128 + tid;
      int krow = cid >> 4;
      int k8 = (cid & 15) * 8;
      *(v8bf*)&Kt[krow * 136 + k8] =
          *(const v8bf*)&qkv[(keyRow0 + krow) * QS + h * 384 + 128 + k8];
    }
    // stage V tile transposed -> Vt[hd][key]; 4-key groups, b64 stores
#pragma unroll
    for (int i = 0; i < 2; ++i) {
      int cid = i * 128 + tid;          // 256 chunks of 4(key) x 8(hd)
      int k4 = (cid >> 4) * 4;          // key group: 0,4,...,60
      int h8 = (cid & 15) * 8;          // hd chunk
      const __bf16* src = &qkv[(keyRow0 + k4) * QS + h * 384 + 256 + h8];
      v8bf r0 = *(const v8bf*)(src);
      v8bf r1 = *(const v8bf*)(src + QS);
      v8bf r2 = *(const v8bf*)(src + 2 * QS);
      v8bf r3 = *(const v8bf*)(src + 3 * QS);
      st_tr4x8(&Vt[h8 * 72 + k4], 72, r0, r1, r2, r3);
    }
    __syncthreads();

    // scores: 16 rows x 64 keys  (Q . K^T)
    v8f Sf[4];
#pragma unroll
    for (int ni = 0; ni < 4; ++ni) {
      v8f acc;
#pragma unroll
      for (int r = 0; r < 8; ++r) acc[r] = 0.f;
#pragma unroll
      for (int c = 0; c < 4; ++c) {
        v16bf Bf = ld_frag(&Kt[(ni * 16 + lh) * 136 + c * 32 + kb]);
        acc = wmma_bf16(Qf[c], Bf, acc);
      }
      Sf[ni] = acc;
    }

    // online softmax; row-wise reductions via 16-lane butterflies (masks 1..8)
    float scale[8];
#pragma unroll
    for (int r = 0; r < 8; ++r) {
      float mx = Sf[0][r];
#pragma unroll
      for (int ni = 1; ni < 4; ++ni) mx = fmaxf(mx, Sf[ni][r]);
#pragma unroll
      for (int m = 1; m <= 8; m <<= 1) mx = fmaxf(mx, __shfl_xor(mx, m, 32));
      float mnew = fmaxf(mrow[r], mx * rscale);
      scale[r] = __expf(mrow[r] - mnew);
      mrow[r] = mnew;
    }

    float rsum[8];
#pragma unroll
    for (int r = 0; r < 8; ++r) rsum[r] = 0.f;
#pragma unroll
    for (int ni = 0; ni < 4; ++ni)
#pragma unroll
      for (int r = 0; r < 8; ++r) {
        float p = __expf(Sf[ni][r] * rscale - mrow[r]);
        Pb[wave][(r + 8 * hb) * 72 + ni * 16 + lh] = f2bf(p);
        rsum[r] += p;
      }
#pragma unroll
    for (int r = 0; r < 8; ++r) {
      float s = rsum[r];
#pragma unroll
      for (int m = 1; m <= 8; m <<= 1) s += __shfl_xor(s, m, 32);
      lrow[r] = lrow[r] * scale[r] + s;
#pragma unroll
      for (int f = 0; f < 8; ++f) Of[f][r] *= scale[r];
    }

    // O += P x V
#pragma unroll
    for (int c2 = 0; c2 < 2; ++c2) {
      v16bf Pf = ld_frag(&Pb[wave][lh * 72 + c2 * 32 + kb]);
#pragma unroll
      for (int f = 0; f < 8; ++f) {
        v16bf Vf = ld_frag(&Vt[(f * 16 + lh) * 72 + c2 * 32 + kb]);
        Of[f] = wmma_bf16(Pf, Vf, Of[f]);
      }
    }
    __syncthreads();
  }

#pragma unroll
  for (int f = 0; f < 8; ++f)
#pragma unroll
    for (int r = 0; r < 8; ++r) {
      long row = rowQ0 + r + 8 * hb;
      float v = Of[f][r] / lrow[r];
      ctx[row * DD + h * HD + f * 16 + lh] = f2bf(v);
    }
}

// ------------------------------------------------------------------ launcher
extern "C" void kernel_launch(void* const* d_in, const int* in_sizes, int n_in,
                              void* d_out, int out_size, void* d_ws, size_t ws_size,
                              hipStream_t stream) {
  (void)in_sizes; (void)n_in; (void)out_size; (void)ws_size;
  const float* hs   = (const float*)d_in[0];
  const float* Wqkv = (const float*)d_in[1];
  const float* bqkv = (const float*)d_in[2];
  const float* Wd   = (const float*)d_in[3];
  const float* bd   = (const float*)d_in[4];
  float* out = (float*)d_out;

  char* ws = (char*)d_ws;
  size_t off = 0;
  __bf16* hsb   = (__bf16*)(ws + off); off += (size_t)BB * SS * DD * 2;
  __bf16* wqkvb = (__bf16*)(ws + off); off += (size_t)DD * 3 * DD * 2;
  __bf16* wdb   = (__bf16*)(ws + off); off += (size_t)DD * DD * 2;
  __bf16* qkvb  = (__bf16*)(ws + off); off += (size_t)BB * SS * 3 * DD * 2;
  __bf16* ctxb  = (__bf16*)(ws + off); off += (size_t)BB * SS * DD * 2;

  long n;
  n = (long)BB * SS * DD;
  cvt_f32_bf16<<<(unsigned)((n / 4 + 255) / 256), 256, 0, stream>>>(hs, hsb, n / 4);
  n = (long)DD * 3 * DD;
  cvt_f32_bf16<<<(unsigned)((n / 4 + 255) / 256), 256, 0, stream>>>(Wqkv, wqkvb, n / 4);
  n = (long)DD * DD;
  cvt_f32_bf16<<<(unsigned)((n / 4 + 255) / 256), 256, 0, stream>>>(Wd, wdb, n / 4);

  // QKV projection: [4096,2048] x [2048,6144] -> bf16 qkv (+bias)
  dim3 g1(3 * DD / 256, BB * SS / 128);
  gemm_bf16_dbuf<true><<<g1, 256, 0, stream>>>(hsb, wqkvb, bqkv, qkvb,
                                               BB * SS, 3 * DD, DD);

  // flash attention -> bf16 ctx
  dim3 ga(SS / 64, HH, BB);
  attn_flash<<<ga, 128, 0, stream>>>(qkvb, ctxb);

  // output projection: [4096,2048] x [2048,2048] -> f32 out (+bias)
  dim3 g2(DD / 256, BB * SS / 128);
  gemm_bf16_dbuf<false><<<g2, 256, 0, stream>>>(ctxb, wdb, bd, out,
                                                BB * SS, DD, DD);
}
